// LSTM_17205638988440
// MI455X (gfx1250) — matrix operand order repaired and checked by reading, not compiled, not run on hardware
//
#include <hip/hip_runtime.h>

typedef __attribute__((ext_vector_type(16))) _Float16 v16h;
typedef __attribute__((ext_vector_type(8)))  _Float16 v8h;
typedef __attribute__((ext_vector_type(8)))  float    v8f;

#define SEQ_LEN 784
#define HIDDEN  128
#define H4      512
#define CLASSES 10
#define ROWS    16      // batch rows per workgroup (one WMMA M-tile)
#define THREADS 256     // 8 wave32 waves

// LDS row pitch (halves) for f16 matrices: 128 + 8 pad -> 272B stride,
// advances LDS bank by 4 per row; 16 lanes cover 64 distinct banks.
#define FPITCH  136

// ---- dynamic LDS layout (byte offsets) ----
// h32  : ROWS*HIDDEN*4      = 8192    @ 0        (final h, f32, for projection)
// xbuf : 2*ROWS*4           = 128     @ 8192     (double-buffered x column)
// whT  : H4*FPITCH*2        = 139264  @ 8320     (B operand, [n][k])
// h16  : 2*ROWS*FPITCH*2    = 8704    @ 147584   (ping-pong A operand)
#define OFF_H32   0
#define OFF_XBUF  8192
#define OFF_WHT   8320
#define OFF_H16   147584
#define SMEM_BYTES 156288

#if __has_builtin(__builtin_amdgcn_tanhf)
__device__ __forceinline__ float fast_tanh(float v) { return __builtin_amdgcn_tanhf(v); }
#else
__device__ __forceinline__ float fast_tanh(float v) { return tanhf(v); }
#endif

// exact identity: sigmoid(x) = 0.5*tanh(x/2) + 0.5  (1 TRANS op)
__device__ __forceinline__ float fast_sigmoid(float v) {
  return __builtin_fmaf(0.5f, fast_tanh(0.5f * v), 0.5f);
}

#define SHUF16(lo, hi) \
  __builtin_shufflevector((lo), (hi), 0,1,2,3,4,5,6,7,8,9,10,11,12,13,14,15)

__global__ void
__launch_bounds__(THREADS, 2)   // 8 waves on one WGP = 2 waves/SIMD: allow big VGPR budget
lstm_persistent_wmma(
    const float* __restrict__ x,
    const float* __restrict__ Wgx, const float* __restrict__ Wgh,
    const float* __restrict__ Wix, const float* __restrict__ Wih,
    const float* __restrict__ Wfx, const float* __restrict__ Wfh,
    const float* __restrict__ Wox, const float* __restrict__ Woh,
    const float* __restrict__ Bg,  const float* __restrict__ Bi,
    const float* __restrict__ Bf,  const float* __restrict__ Bo,
    const float* __restrict__ Wph, const float* __restrict__ Bp,
    float* __restrict__ out)
{
  extern __shared__ char smem[];
  float*    h32  = (float*)(smem + OFF_H32);
  float*    xbuf = (float*)(smem + OFF_XBUF);     // [2][16]
  _Float16* whT  = (_Float16*)(smem + OFF_WHT);   // [512][FPITCH]
  _Float16* h16  = (_Float16*)(smem + OFF_H16);   // [2][16][FPITCH]

  const int tid  = threadIdx.x;
  const int lane = tid & 31;
  const int wv   = tid >> 5;       // wave id 0..7
  const int hi16 = lane >> 4;      // 0: lanes 0-15, 1: lanes 16-31
  const int l15  = lane & 15;
  const int bbase = blockIdx.x * ROWS;

  // ---------------- one-time preload ----------------
  // Wh^T in f16: whT[n][k] = W?h[k][n&127], gate blocks concatenated (g,i,f,o)
  for (int idx = tid; idx < H4 * HIDDEN; idx += THREADS) {
    int n = idx >> 7, k = idx & 127;
    int blk = n >> 7, nn = n & 127;
    const float* W = (blk == 0) ? Wgh : (blk == 1) ? Wih : (blk == 2) ? Wfh : Woh;
    whT[n * FPITCH + k] = (_Float16)W[k * HIDDEN + nn];
  }
  for (int idx = tid; idx < 2 * ROWS * HIDDEN; idx += THREADS) {
    int buf = idx >> 11, r = (idx >> 7) & 15, k = idx & 127;
    h16[buf * ROWS * FPITCH + r * FPITCH + k] = (_Float16)0.0f;
  }
  if (tid < ROWS) xbuf[tid] = x[(bbase + tid) * SEQ_LEN + 0];  // x column for t=0
  __syncthreads();

  // ---------------- hoist loop-invariants into registers ----------------
  // This wave owns column nblk = wv*16 + l15 of EVERY gate block (tiles
  // wv, wv+8, wv+16, wv+24), so gates mix entirely within one lane.
  const int nblk = wv * 16 + l15;
  const float wxg = Wgx[nblk], wxi = Wix[nblk], wxf = Wfx[nblk], wxo = Wox[nblk];
  const float bgs = Bg[nblk],  bis = Bi[nblk],  bfs = Bf[nblk],  bos = Bo[nblk];

  // B fragments: 4 gate blocks x 4 K-chunks, resident in VGPRs for all steps.
  // Layout used here, contiguous k (16-bit B 32x16): lanes 0-15 K=c*32+0..15, lanes 16-31 +16;
  // N = lane&15. Contiguous in k in our transposed whT.
  v16h bfrag[4][4];
#pragma unroll
  for (int gi = 0; gi < 4; ++gi) {
    const _Float16* bb = whT + ((gi * 8 + wv) * 16 + l15) * FPITCH;
#pragma unroll
    for (int kc = 0; kc < 4; ++kc) {
      const _Float16* q = bb + kc * 32 + hi16 * 16;
      v8h blo = *(const v8h*)q;
      v8h bhi = *(const v8h*)(q + 8);
      bfrag[gi][kc] = SHUF16(blo, bhi);
    }
  }

  // Cell state c for this lane's 8 (row,col) elements lives in registers.
  float c_reg[8] = {0,0,0,0,0,0,0,0};
  float hlast[8] = {0,0,0,0,0,0,0,0};

  // ---------------- 784-step recurrence (one barrier per step) ----------------
  for (int t = 0; t < SEQ_LEN; ++t) {
    const _Float16* hcur = h16 + (t & 1) * ROWS * FPITCH;
    _Float16*       hnxt = h16 + ((t + 1) & 1) * ROWS * FPITCH;

    // kc-outer / gate-inner: only ONE A fragment (8 VGPRs) live at a time.
    v8f acc[4];
    {
      v8f zero = {};
      acc[0] = zero; acc[1] = zero; acc[2] = zero; acc[3] = zero;
    }
#pragma unroll
    for (int kc = 0; kc < 4; ++kc) {
      // A fragment (16x32 chunk of h). Layout used (contiguous k): lane row M=lane&15;
      // lanes 0-15 K=c*32+0..15, lanes 16-31 K=c*32+16..31 (matches B).
      const _Float16* p = hcur + l15 * FPITCH + kc * 32 + hi16 * 16;
      v8h lo = *(const v8h*)p;
      v8h hi = *(const v8h*)(p + 8);
      v16h a = SHUF16(lo, hi);
#pragma unroll
      for (int gi = 0; gi < 4; ++gi) {
        acc[gi] = __builtin_amdgcn_wmma_f32_16x16x32_f16(
            false, a, false, bfrag[gi][kc], (short)0, acc[gi], false, false);
      }
    }

    // Stage next step's x column while WMMAs drain.
    if (tid < ROWS) {
      int tn = (t + 1 < SEQ_LEN) ? (t + 1) : t;
      xbuf[((t + 1) & 1) * ROWS + tid] = x[(bbase + tid) * SEQ_LEN + tn];
      int tp = (t + 8 < SEQ_LEN) ? (t + 8) : t;
      __builtin_prefetch(&x[(bbase + tid) * SEQ_LEN + tp], 0, 3);
    }

    // Gates fully in registers. C/D layout: VGPR r -> row m = r + 8*hi16.
    const float* xp = xbuf + (t & 1) * ROWS + 8 * hi16;
#pragma unroll
    for (int r = 0; r < 8; ++r) {
      float xv = xp[r];
      float zg = acc[0][r] + xv * wxg + bgs;
      float zi = acc[1][r] + xv * wxi + bis;
      float zf = acc[2][r] + xv * wxf + bfs;
      float zo = acc[3][r] + xv * wxo + bos;
      float g  = fast_tanh(zg);
      float ig = fast_sigmoid(zi);
      float fg = fast_sigmoid(zf);
      float og = fast_sigmoid(zo);
      float cn = g * ig + c_reg[r] * fg;
      float hn = fast_tanh(cn * og);          // reference: h = tanh(c*o)
      c_reg[r] = cn;
      hlast[r] = hn;
      hnxt[(r + 8 * hi16) * FPITCH + nblk] = (_Float16)hn;
    }
    __syncthreads();
  }

  // ---------------- final projection: h @ Wph + Bp ----------------
#pragma unroll
  for (int r = 0; r < 8; ++r)
    h32[(r + 8 * hi16) * HIDDEN + nblk] = hlast[r];
  __syncthreads();

  if (tid < ROWS * CLASSES) {
    int b = tid / CLASSES, cls = tid % CLASSES;
    float s = Bp[cls];
#pragma unroll 8
    for (int k = 0; k < HIDDEN; ++k)
      s += h32[b * HIDDEN + k] * Wph[k * CLASSES + cls];
    out[(bbase + b) * CLASSES + cls] = s;
  }
}

extern "C" void kernel_launch(void* const* d_in, const int* in_sizes, int n_in,
                              void* d_out, int out_size, void* d_ws, size_t ws_size,
                              hipStream_t stream) {
  (void)in_sizes; (void)n_in; (void)d_ws; (void)ws_size; (void)out_size;
  const float* x   = (const float*)d_in[0];
  const float* Wgx = (const float*)d_in[1];
  const float* Wgh = (const float*)d_in[2];
  const float* Wix = (const float*)d_in[3];
  const float* Wih = (const float*)d_in[4];
  const float* Wfx = (const float*)d_in[5];
  const float* Wfh = (const float*)d_in[6];
  const float* Wox = (const float*)d_in[7];
  const float* Woh = (const float*)d_in[8];
  const float* Bg  = (const float*)d_in[9];
  const float* Bi  = (const float*)d_in[10];
  const float* Bf  = (const float*)d_in[11];
  const float* Bo  = (const float*)d_in[12];
  const float* Wph = (const float*)d_in[13];
  const float* Bp  = (const float*)d_in[14];
  float* out = (float*)d_out;

  (void)hipFuncSetAttribute((const void*)lstm_persistent_wmma,
                            hipFuncAttributeMaxDynamicSharedMemorySize,
                            SMEM_BYTES);

  // 256 batch rows / 16 rows per WG = 16 persistent workgroups, 8 waves each.
  lstm_persistent_wmma<<<dim3(256 / ROWS), dim3(THREADS), SMEM_BYTES, stream>>>(
      x, Wgx, Wgh, Wix, Wih, Wfx, Wfh, Wox, Woh,
      Bg, Bi, Bf, Bo, Wph, Bp, out);
}
